// TransformerModule_36756330119313
// MI455X (gfx1250) — compile-verified, matrix operand order checked by reading, not run on hardware
//
#include <hip/hip_runtime.h>
#include <hip/hip_bf16.h>

// ---------------------------------------------------------------------------
// Types for CDNA5 WMMA (wave32): A/B = 16 x bf16 (8 VGPRs), C/D = 8 x f32.
// ---------------------------------------------------------------------------
typedef __attribute__((ext_vector_type(16))) __bf16 v16bf;
typedef __attribute__((ext_vector_type(8)))  __bf16 v8bf;
typedef __attribute__((ext_vector_type(8)))  float  v8f;

__device__ __forceinline__ __bf16 f2bf(float f) {
  union { float f; unsigned u; } a; a.f = f;
  unsigned r = a.u + 0x7FFFu + ((a.u >> 16) & 1u);   // round-to-nearest-even
  union { unsigned short s; __bf16 b; } o; o.s = (unsigned short)(r >> 16);
  return o.b;
}

__device__ __forceinline__ float gelu_f(float v) {
  return 0.5f * v * (1.0f + erff(v * 0.70710678118654752f));
}

// ---------------------------------------------------------------------------
// Transpose + fp32->bf16 convert:  in[R,C] (f32, row-major) -> out[C,R] (bf16)
// ---------------------------------------------------------------------------
__global__ __launch_bounds__(256) void transpose_to_bf16(
    const float* __restrict__ in, __bf16* __restrict__ out, int R, int C) {
  __shared__ float tile[32][33];
  int bx = blockIdx.x * 32, by = blockIdx.y * 32;
  int tx = threadIdx.x, ty = threadIdx.y;               // block (32, 8)
#pragma unroll
  for (int i = 0; i < 32; i += 8)
    tile[ty + i][tx] = in[(size_t)(by + ty + i) * C + (bx + tx)];
  __syncthreads();
#pragma unroll
  for (int i = 0; i < 32; i += 8)
    out[(size_t)(bx + ty + i) * R + (by + tx)] = f2bf(tile[tx][ty + i]);
}

// ---------------------------------------------------------------------------
// LayerNorm (D=1024) fp32 -> bf16 : one block (256 threads) per row.
// ---------------------------------------------------------------------------
__global__ __launch_bounds__(256) void layernorm_to_bf16(
    const float* __restrict__ x, const float* __restrict__ w,
    const float* __restrict__ b, __bf16* __restrict__ t) {
  const int D_ = 1024;
  size_t row = blockIdx.x;
  const float* xr = x + row * D_;
  float v[4]; float s = 0.f, ss = 0.f;
#pragma unroll
  for (int i = 0; i < 4; ++i) {
    v[i] = xr[threadIdx.x + 256 * i];
    s += v[i]; ss += v[i] * v[i];
  }
#pragma unroll
  for (int off = 1; off < 32; off <<= 1) {              // wave32 reduce
    s  += __shfl_xor(s,  off);
    ss += __shfl_xor(ss, off);
  }
  __shared__ float rs[8], rss[8];
  int wv = threadIdx.x >> 5, ln = threadIdx.x & 31;
  if (ln == 0) { rs[wv] = s; rss[wv] = ss; }
  __syncthreads();
  s = 0.f; ss = 0.f;
#pragma unroll
  for (int i = 0; i < 8; ++i) { s += rs[i]; ss += rss[i]; }
  float mean = s * (1.f / 1024.f);
  float var  = ss * (1.f / 1024.f) - mean * mean;
  float rstd = rsqrtf(var + 1e-5f);
#pragma unroll
  for (int i = 0; i < 4; ++i) {
    int c = threadIdx.x + 256 * i;
    t[row * D_ + c] = f2bf((v[i] - mean) * rstd * w[c] + b[c]);
  }
}

// ---------------------------------------------------------------------------
// GEMM: C[M,Nn] = epilogue( A[M,K](bf16) x Bt[Nn,K](bf16)^T + bias ) (+resid)
// Block = 256 threads = 8 waves; macro-tile 128x128, BK=32; wave tile 32x64.
// Double-buffered LDS, register prefetch, single barrier per k-step.
// ---------------------------------------------------------------------------
template<bool GELU, bool RESID, bool OUTBF>
__global__ __launch_bounds__(256) void gemm_bf16_wmma(
    const __bf16* __restrict__ A,  int lda,
    const __bf16* __restrict__ Bt, int ldb,
    const float*  __restrict__ bias,
    const float*  __restrict__ resid,
    __bf16* __restrict__ outB, float* __restrict__ outF,
    int Nn, int K) {
  __shared__ __bf16 As[2][128 * 32];   // 2 x 8 KB
  __shared__ __bf16 Bs[2][128 * 32];   // 2 x 8 KB

  const int tid  = threadIdx.x;
  const int wave = tid >> 5;
  const int lane = tid & 31;
  const int half = lane >> 4;
  const int l16  = lane & 15;
  const int kb   = half * 8;
  const int wm   = wave >> 1;          // 0..3 : 32-row band
  const int wn   = wave & 1;           // 0..1 : 64-col band
  const int rowBase = blockIdx.y * 128;
  const int colBase = blockIdx.x * 128;

  v8f acc[2][4];
#pragma unroll
  for (int rt = 0; rt < 2; ++rt)
#pragma unroll
    for (int c = 0; c < 4; ++c)
#pragma unroll
      for (int j = 0; j < 8; ++j) acc[rt][c][j] = 0.f;

  // per-thread staging: 2 b128 segs of A + 2 of B per k-step
  v8bf pa[2], pb[2];
  auto fetch = [&](int k0) {
#pragma unroll
    for (int r = 0; r < 2; ++r) {
      int id = tid + 256 * r, rw = id >> 2, sg = id & 3;
      pa[r] = *(const v8bf*)&A [(size_t)(rowBase + rw) * lda + k0 + sg * 8];
      pb[r] = *(const v8bf*)&Bt[(size_t)(colBase + rw) * ldb + k0 + sg * 8];
    }
  };
  auto stash = [&](int buf) {
#pragma unroll
    for (int r = 0; r < 2; ++r) {
      int id = tid + 256 * r, rw = id >> 2, sg = id & 3;
      *(v8bf*)&As[buf][rw * 32 + sg * 8] = pa[r];
      *(v8bf*)&Bs[buf][rw * 32 + sg * 8] = pb[r];
    }
  };

  const int nIter = K >> 5;
  fetch(0);
  stash(0);
  if (nIter > 1) fetch(32);

  for (int it = 0; it < nIter; ++it) {
    __syncthreads();                       // single barrier per k-step
    const int buf = it & 1;
    if (it + 1 < nIter) {
      stash(buf ^ 1);                      // tile it+1 -> other buffer
      if (it + 2 < nIter) fetch((it + 2) * 32);  // overlap with WMMAs below
    }

    // load all fragments first, then the WMMA burst
    v16bf a[2], bfr[4];
#pragma unroll
    for (int rt = 0; rt < 2; ++rt) {
      int arow = wm * 32 + rt * 16 + l16;
      ((v8bf*)&a[rt])[0] = *(const v8bf*)&As[buf][arow * 32 + kb];
      ((v8bf*)&a[rt])[1] = *(const v8bf*)&As[buf][arow * 32 + kb + 16];
    }
#pragma unroll
    for (int c = 0; c < 4; ++c) {
      int brow = wn * 64 + c * 16 + l16;
      ((v8bf*)&bfr[c])[0] = *(const v8bf*)&Bs[buf][brow * 32 + kb];
      ((v8bf*)&bfr[c])[1] = *(const v8bf*)&Bs[buf][brow * 32 + kb + 16];
    }
#pragma unroll
    for (int rt = 0; rt < 2; ++rt)
#pragma unroll
      for (int c = 0; c < 4; ++c)
        acc[rt][c] = __builtin_amdgcn_wmma_f32_16x16x32_bf16(
            false, a[rt], false, bfr[c], (short)0, acc[rt][c], false, false);
  }

  // epilogue (C/D layout: VGPR j -> row j + 8*half, lane&15 -> column)
#pragma unroll
  for (int rt = 0; rt < 2; ++rt) {
    int row0 = rowBase + wm * 32 + rt * 16 + half * 8;
#pragma unroll
    for (int c = 0; c < 4; ++c) {
      int col = colBase + wn * 64 + c * 16 + l16;
      float bv = bias ? bias[col] : 0.f;
#pragma unroll
      for (int j = 0; j < 8; ++j) {
        size_t idx = (size_t)(row0 + j) * Nn + col;
        float v = acc[rt][c][j] + bv;
        if (GELU)  v = gelu_f(v);
        if (RESID) v += resid[idx];
        if (OUTBF) outB[idx] = f2bf(v);
        else       outF[idx] = v;
      }
    }
  }
}

// ---------------------------------------------------------------------------
// Flash attention: block = 128 threads = 4 waves, one (b,h,64-row Q block).
// Each wave owns 16 Q rows; K chunk (32x64) row-major + V chunk transposed
// (64x32) staged in LDS; online softmax with 16-lane shfl reductions; P goes
// through per-wave LDS to convert C-layout -> A-layout for the PV matmul.
// ---------------------------------------------------------------------------
__global__ __launch_bounds__(128) void attention_kernel(
    const __bf16* __restrict__ qkv,   // [B*N, 3*INNER]
    __bf16* __restrict__ O) {         // [B*N, INNER]
  const int Nn = 2048, HD_ = 64, QLD = 3072, INNER_ = 1024;
  const float scale = 0.125f;         // 64^-0.5

  int blk = blockIdx.x;
  int rb  = blk & 31;                 // N/64 row-blocks
  int bh  = blk >> 5;
  int b_  = bh >> 4;
  int h_  = bh & 15;
  int tid  = threadIdx.x;
  int wv   = tid >> 5;
  int lane = tid & 31;
  int half = lane >> 4;
  int l16  = lane & 15;
  int kb   = half * 8;

  size_t rowb = (size_t)b_ * Nn;
  const __bf16* qp = qkv + rowb * QLD + h_ * HD_;
  const __bf16* kp = qp + INNER_;
  const __bf16* vp = qp + 2 * INNER_;

  __shared__ __bf16 Ks[32 * 64];      // K chunk, [key][dim]
  __shared__ __bf16 Vt[64 * 32];      // V chunk transposed, [dim][key]
  __shared__ __bf16 Ps[4][16 * 32];   // per-wave P tile

  // Q fragments for this wave's 16 rows (two 16x32 k-steps over HD=64)
  int qrow = rb * 64 + wv * 16 + l16;
  v16bf qa[2];
#pragma unroll
  for (int s2 = 0; s2 < 2; ++s2) {
    ((v8bf*)&qa[s2])[0] = *(const v8bf*)&qp[(size_t)qrow * QLD + s2 * 32 + kb];
    ((v8bf*)&qa[s2])[1] = *(const v8bf*)&qp[(size_t)qrow * QLD + s2 * 32 + kb + 16];
  }

  float mrow[8], lrow[8];
  v8f o[4];
#pragma unroll
  for (int j = 0; j < 8; ++j) { mrow[j] = -3.0e38f; lrow[j] = 0.f; }
#pragma unroll
  for (int c = 0; c < 4; ++c)
#pragma unroll
    for (int j = 0; j < 8; ++j) o[c][j] = 0.f;

#pragma unroll 1
  for (int j0 = 0; j0 < Nn; j0 += 32) {
    __syncthreads();
    // stage K (row-major) and V (transposed)
#pragma unroll
    for (int s2 = 0; s2 < 2; ++s2) {
      int id = tid + 128 * s2;
      int key = id >> 3, part = id & 7;
      *(v8bf*)&Ks[key * 64 + part * 8] =
          *(const v8bf*)&kp[(size_t)(j0 + key) * QLD + part * 8];
      v8bf vv = *(const v8bf*)&vp[(size_t)(j0 + key) * QLD + part * 8];
#pragma unroll
      for (int e = 0; e < 8; ++e) Vt[(part * 8 + e) * 32 + key] = vv[e];
    }
    __syncthreads();

    // S[16 x 32] = Q @ K^T  (two 16-key column tiles)
    v8f s0, s1;
#pragma unroll
    for (int j = 0; j < 8; ++j) { s0[j] = 0.f; s1[j] = 0.f; }
#pragma unroll
    for (int s2 = 0; s2 < 2; ++s2) {
      v16bf b0, b1;
      ((v8bf*)&b0)[0] = *(const v8bf*)&Ks[l16 * 64 + s2 * 32 + kb];
      ((v8bf*)&b0)[1] = *(const v8bf*)&Ks[l16 * 64 + s2 * 32 + kb + 16];
      ((v8bf*)&b1)[0] = *(const v8bf*)&Ks[(16 + l16) * 64 + s2 * 32 + kb];
      ((v8bf*)&b1)[1] = *(const v8bf*)&Ks[(16 + l16) * 64 + s2 * 32 + kb + 16];
      s0 = __builtin_amdgcn_wmma_f32_16x16x32_bf16(
          false, qa[s2], false, b0, (short)0, s0, false, false);
      s1 = __builtin_amdgcn_wmma_f32_16x16x32_bf16(
          false, qa[s2], false, b1, (short)0, s1, false, false);
    }

    // online softmax (row = j + 8*half; row stats reduced over 16 lanes)
#pragma unroll
    for (int j = 0; j < 8; ++j) {
      float a0 = s0[j] * scale, a1 = s1[j] * scale;
      float cm = fmaxf(a0, a1);
#pragma unroll
      for (int off = 1; off < 16; off <<= 1) cm = fmaxf(cm, __shfl_xor(cm, off));
      float mn = fmaxf(mrow[j], cm);
      float alpha = __expf(mrow[j] - mn);
      float p0 = __expf(a0 - mn), p1 = __expf(a1 - mn);
      float rsum = p0 + p1;
#pragma unroll
      for (int off = 1; off < 16; off <<= 1) rsum += __shfl_xor(rsum, off);
      lrow[j] = lrow[j] * alpha + rsum;
      mrow[j] = mn;
#pragma unroll
      for (int c = 0; c < 4; ++c) o[c][j] *= alpha;
      Ps[wv][(j + 8 * half) * 32 + l16]      = f2bf(p0);
      Ps[wv][(j + 8 * half) * 32 + 16 + l16] = f2bf(p1);
    }
    __syncthreads();   // P C-layout -> A-layout visibility across lanes

    // O += P @ V
    v16bf pa;
    ((v8bf*)&pa)[0] = *(const v8bf*)&Ps[wv][l16 * 32 + kb];
    ((v8bf*)&pa)[1] = *(const v8bf*)&Ps[wv][l16 * 32 + kb + 16];
#pragma unroll
    for (int c = 0; c < 4; ++c) {
      v16bf vb;
      ((v8bf*)&vb)[0] = *(const v8bf*)&Vt[(c * 16 + l16) * 32 + kb];
      ((v8bf*)&vb)[1] = *(const v8bf*)&Vt[(c * 16 + l16) * 32 + kb + 16];
      o[c] = __builtin_amdgcn_wmma_f32_16x16x32_bf16(
          false, pa, false, vb, (short)0, o[c], false, false);
    }
  }

#pragma unroll
  for (int c = 0; c < 4; ++c)
#pragma unroll
    for (int j = 0; j < 8; ++j) {
      int orow = rb * 64 + wv * 16 + j + 8 * half;
      int ocol = h_ * HD_ + c * 16 + l16;
      O[(rowb + orow) * INNER_ + ocol] = f2bf(o[c][j] * (1.f / lrow[j]));
    }
}

// ---------------------------------------------------------------------------
// Launcher
// ---------------------------------------------------------------------------
extern "C" void kernel_launch(void* const* d_in, const int* in_sizes, int n_in,
                              void* d_out, int out_size, void* d_ws, size_t ws_size,
                              hipStream_t stream) {
  (void)in_sizes; (void)n_in; (void)out_size; (void)ws_size;
  const float* x    = (const float*)d_in[0];
  const float* ln_w = (const float*)d_in[1];
  const float* ln_b = (const float*)d_in[2];
  const float* wqkv = (const float*)d_in[3];
  const float* wo   = (const float*)d_in[4];
  const float* bo   = (const float*)d_in[5];
  const float* w1   = (const float*)d_in[6];
  const float* b1   = (const float*)d_in[7];
  const float* w2   = (const float*)d_in[8];
  const float* b2   = (const float*)d_in[9];
  float* xbuf = (float*)d_out;            // fp32 residual stream lives in d_out

  constexpr int B_ = 4, N_ = 2048, D_ = 1024, INNER_ = 1024, MLP_ = 4096,
                L_ = 4, H_ = 16;
  const size_t M = (size_t)B_ * N_;       // 8192 rows

  char* wsp = (char*)d_ws;
  auto alloc = [&](size_t bytes) {
    char* p = wsp; wsp += (bytes + 255) & ~(size_t)255; return p;
  };
  __bf16* wqkvT = (__bf16*)alloc((size_t)L_ * 3 * INNER_ * D_ * 2);
  __bf16* woT   = (__bf16*)alloc((size_t)L_ * D_ * INNER_ * 2);
  __bf16* w1T   = (__bf16*)alloc((size_t)L_ * MLP_ * D_ * 2);
  __bf16* w2T   = (__bf16*)alloc((size_t)L_ * D_ * MLP_ * 2);
  __bf16* tb    = (__bf16*)alloc(M * D_ * 2);
  __bf16* qkvb  = (__bf16*)alloc(M * 3 * INNER_ * 2);
  __bf16* Ob    = (__bf16*)alloc(M * INNER_ * 2);
  __bf16* hb    = (__bf16*)alloc(M * MLP_ * 2);

  // residual stream = input x
  hipMemcpyAsync(xbuf, x, M * D_ * sizeof(float), hipMemcpyDeviceToDevice, stream);

  // one-time per-launch weight transpose + bf16 convert
  dim3 tblk(32, 8);
  for (int i = 0; i < L_; ++i) {
    transpose_to_bf16<<<dim3(3 * INNER_ / 32, D_ / 32), tblk, 0, stream>>>(
        wqkv + (size_t)i * D_ * 3 * INNER_, wqkvT + (size_t)i * 3 * INNER_ * D_,
        D_, 3 * INNER_);
    transpose_to_bf16<<<dim3(D_ / 32, INNER_ / 32), tblk, 0, stream>>>(
        wo + (size_t)i * INNER_ * D_, woT + (size_t)i * D_ * INNER_,
        INNER_, D_);
    transpose_to_bf16<<<dim3(MLP_ / 32, D_ / 32), tblk, 0, stream>>>(
        w1 + (size_t)i * D_ * MLP_, w1T + (size_t)i * MLP_ * D_, D_, MLP_);
    transpose_to_bf16<<<dim3(D_ / 32, MLP_ / 32), tblk, 0, stream>>>(
        w2 + (size_t)i * MLP_ * D_, w2T + (size_t)i * D_ * MLP_, MLP_, D_);
  }

  for (int i = 0; i < L_; ++i) {
    const __bf16* wqkvT_i = wqkvT + (size_t)i * 3 * INNER_ * D_;
    const __bf16* woT_i   = woT   + (size_t)i * D_ * INNER_;
    const __bf16* w1T_i   = w1T   + (size_t)i * MLP_ * D_;
    const __bf16* w2T_i   = w2T   + (size_t)i * D_ * MLP_;

    // 1) t = LN(x)
    layernorm_to_bf16<<<(int)M, 256, 0, stream>>>(xbuf, ln_w, ln_b, tb);
    // 2) qkv = t @ wqkv[i]
    gemm_bf16_wmma<false, false, true><<<dim3(3072 / 128, 8192 / 128), 256, 0, stream>>>(
        tb, D_, wqkvT_i, D_, nullptr, nullptr, qkvb, nullptr, 3 * INNER_, D_);
    // 3) O = attention(q,k,v)
    attention_kernel<<<B_ * H_ * (N_ / 64), 128, 0, stream>>>(qkvb, Ob);
    // 4) x = x + O @ wo[i] + bo[i]
    gemm_bf16_wmma<false, true, false><<<dim3(1024 / 128, 8192 / 128), 256, 0, stream>>>(
        Ob, INNER_, woT_i, INNER_, bo + (size_t)i * D_, xbuf, nullptr, xbuf, D_, INNER_);
    // 5) t = LN(x)
    layernorm_to_bf16<<<(int)M, 256, 0, stream>>>(xbuf, ln_w, ln_b, tb);
    // 6) h = gelu(t @ w1[i] + b1[i])
    gemm_bf16_wmma<true, false, true><<<dim3(4096 / 128, 8192 / 128), 256, 0, stream>>>(
        tb, D_, w1T_i, D_, b1 + (size_t)i * MLP_, nullptr, hb, nullptr, MLP_, D_);
    // 7) x = x + gelu(h @ w2[i] + b2[i])
    gemm_bf16_wmma<true, true, false><<<dim3(1024 / 128, 8192 / 128), 256, 0, stream>>>(
        hb, MLP_, w2T_i, MLP_, b2 + (size_t)i * D_, xbuf, nullptr, xbuf, D_, MLP_);
  }
}
